// GSPNBlock_12807592477361
// MI455X (gfx1250) — matrix operand order repaired and checked by reading, not compiled, not run on hardware
//
#include <hip/hip_runtime.h>
#include <hip/hip_bf16.h>
#include <math.h>

// ---------------------------------------------------------------------------
// GSPN block for MI455X (gfx1250, wave32, WMMA).
// bf16 datapath with f32 accumulation via v_wmma_f32_16x16x32_bf16.
// B-operand tiles staged in LDS via GLOBAL_LOAD_ASYNC_TO_LDS_B128 (ASYNCcnt);
// synchronous b128 copies as fallback if the builtin is unavailable.
// ---------------------------------------------------------------------------

typedef __bf16 bf16;
typedef __attribute__((ext_vector_type(16))) __bf16 v16bf;
typedef __attribute__((ext_vector_type(8)))  float  v8f;
typedef __attribute__((ext_vector_type(4)))  unsigned int u32x4;
typedef int i32x4v __attribute__((vector_size(16)));   // matches builtin param type

union V16 { v16bf v; u32x4 u[2]; };

#if __has_builtin(__builtin_amdgcn_global_load_async_to_lds_b128) && \
    __has_builtin(__builtin_amdgcn_s_wait_asynccnt)
#define HAVE_ASYNC_LDS 1
// 16-byte async global->LDS copy (ASYNCcnt-tracked).
__device__ __forceinline__ void async_copy16(const bf16* g, bf16* l) {
    __builtin_amdgcn_global_load_async_to_lds_b128(
        (__attribute__((address_space(1))) i32x4v*)(unsigned long long)(g),
        (__attribute__((address_space(3))) i32x4v*)(unsigned int)(unsigned long long)(l),
        0, 0);
}
#else
#define HAVE_ASYNC_LDS 0
#endif

#define DIM      256
#define BATCH    8
#define HDIM     64
#define WDIM     64
#define S_PER_B  4096                 // H*W
#define OC_TOTAL (17 * DIM)           // 4352 fused pointwise output channels
#define OFF_U    0
#define OFF_W    (DIM)                // 256  : wraw starts here (12*C channels)
#define OFF_L    (13 * DIM)           // 3328 : lam-raw starts here (4*C channels)

// bf16-weight arena layout (element offsets)
#define WT_WU   0
#define WT_WW   (WT_WU  + DIM * DIM)        // Wu rows o=0..255       (o,c)
#define WT_WL   (WT_WW  + 12 * DIM * DIM)   // Ww rows o=256..3327    (o,c)
#define WT_WO   (WT_WL  + 4 * DIM * DIM)    // Wl rows o=3328..4351   (o,c)
#define WT_W1T  (WT_WO  + DIM * DIM)        // Wo (o,c)
#define WT_W2T  (WT_W1T + 4 * DIM * DIM)    // W1^T (j,c) 1024x256
#define WT_TOTAL (WT_W2T + 4 * DIM * DIM)   // W2^T (o,j) 256x1024

// ---------------------------------------------------------------------------
// Kernel 0: weights -> bf16 (transposing W1,W2 so K is row-contiguous) and
//           concatenated f32 bias vector bias_all = [bu | bw | bl] (4352).
// ---------------------------------------------------------------------------
__global__ void k_prep_weights(const float* __restrict__ Wu, const float* __restrict__ Ww,
                               const float* __restrict__ Wl, const float* __restrict__ Wo,
                               const float* __restrict__ W1, const float* __restrict__ W2,
                               const float* __restrict__ bu, const float* __restrict__ bw,
                               const float* __restrict__ bl,
                               bf16* __restrict__ wt, float* __restrict__ bias_all) {
    int i = blockIdx.x * 256 + threadIdx.x;
    if (i < DIM * DIM)      wt[WT_WU + i] = (bf16)Wu[i];
    if (i < 12 * DIM * DIM) wt[WT_WW + i] = (bf16)Ww[i];
    if (i < 4 * DIM * DIM)  wt[WT_WL + i] = (bf16)Wl[i];
    if (i < DIM * DIM)      wt[WT_WO + i] = (bf16)Wo[i];
    if (i < 4 * DIM * DIM) {            // W1 is (c=256, j=1024); store (j,c)
        int j = i / DIM, c = i % DIM;
        wt[WT_W1T + i] = (bf16)W1[c * (4 * DIM) + j];
    }
    if (i < 4 * DIM * DIM) {            // W2 is (j=1024, o=256); store (o,j)
        int o = i / (4 * DIM), j = i % (4 * DIM);
        wt[WT_W2T + i] = (bf16)W2[j * DIM + o];
    }
    if (i < DIM)            bias_all[OFF_U + i] = bu[i];
    if (i < 12 * DIM)       bias_all[OFF_W + i] = bw[i];
    if (i < 4 * DIM)        bias_all[OFF_L + i] = bl[i];
}

// ---------------------------------------------------------------------------
// Kernel 1: LDS-tiled transpose  x[b][c][s] -> xbf_t[b][s][c] (bf16),
//           zero the scan accumulator y. Coalesced reads and 128-bit writes.
// ---------------------------------------------------------------------------
__global__ void __launch_bounds__(256) k_prep_x(const float* __restrict__ x,
                                                bf16* __restrict__ xbf_t,
                                                float* __restrict__ y) {
    __shared__ __align__(16) bf16 T[64][72];
    const int b = blockIdx.z, c0 = blockIdx.x * 64, s0 = blockIdx.y * 64;
    for (int idx = threadIdx.x; idx < 64 * 64; idx += 256) {
        int cl = idx >> 6, sl = idx & 63;
        size_t gi = ((size_t)b * DIM + c0 + cl) * S_PER_B + s0 + sl;
        T[sl][cl] = (bf16)x[gi];
        y[gi] = 0.0f;
    }
    __syncthreads();
    for (int idx = threadIdx.x; idx < 64 * 8; idx += 256) {
        int sl = idx >> 3, cb = (idx & 7) * 8;
        *(u32x4*)(&xbf_t[((size_t)b * S_PER_B + s0 + sl) * DIM + c0 + cb]) =
            *(const u32x4*)(&T[sl][cb]);
    }
}

// ---------------------------------------------------------------------------
// Kernel 2: fused pointwise GEMM  pw[b][o][s] = W[o,:] . x[b,:,s] + bias
//   M = 4352 out channels, N = 32768 positions, K = 256, bf16->f32 WMMA.
//   Block: 256 thr (8 waves) -> tile 64M x 32N.
//   B tile (x, pre-transposed [s][c]) staged into LDS with async b128 copies.
// ---------------------------------------------------------------------------
__global__ void __launch_bounds__(256) k_pointwise(
        const bf16* __restrict__ wt, const bf16* __restrict__ xbf_t,
        const float* __restrict__ bias_all, bf16* __restrict__ pw) {
    __shared__ __align__(16) bf16 Bs[32][264];   // [n_local][k], padded row

    const int b    = blockIdx.z;
    const int mblk = blockIdx.x * 64;
    const int nblk = blockIdx.y * 32;
    const int tid  = threadIdx.x;

    // Stage 32 rows x 512B (tile s=nblk..+31, c=0..255) -> Bs[s][c]
    const bf16* gsrc = xbf_t + ((size_t)b * S_PER_B + nblk) * DIM;
#if HAVE_ASYNC_LDS
    for (int it = 0; it < 4; ++it) {
        int chunk = it * 256 + tid;          // 1024 x 16B chunks
        int r = chunk >> 5, cc = (chunk & 31) * 8;
        async_copy16(gsrc + (size_t)r * DIM + cc, &Bs[r][cc]);
    }
    __builtin_amdgcn_s_wait_asynccnt(0);
#else
    for (int it = 0; it < 4; ++it) {
        int chunk = it * 256 + tid;
        int r = chunk >> 5, cc = (chunk & 31) * 8;
        *(u32x4*)(&Bs[r][cc]) = *(const u32x4*)(gsrc + (size_t)r * DIM + cc);
    }
#endif
    __syncthreads();

    const int wave = tid >> 5, lane = tid & 31;
    const int wm = wave >> 1, wn = wave & 1;
    const int half = lane >> 4, l16 = lane & 15;
    const int m0 = mblk + wm * 16;

    const bf16* Arow = wt + (size_t)(m0 + l16) * DIM;   // rows o contiguous across Wu|Ww|Wl
    const bf16* Brow = &Bs[wn * 16 + l16][0];

    v8f acc = {};
    for (int kb = 0; kb < DIM; kb += 32) {
        V16 a, bb;
        int ka = kb + half * 8;              // A lane layout: K chunks {ka..+7, ka+16..+23}
        a.u[0] = *(const u32x4*)(Arow + ka);
        a.u[1] = *(const u32x4*)(Arow + ka + 16);
        int k2 = kb + half * 8;              // B lane layout: K chunks {k2..+7, k2+16..+23}
        bb.u[0] = *(const u32x4*)(Brow + k2);
        bb.u[1] = *(const u32x4*)(Brow + k2 + 16);
        acc = __builtin_amdgcn_wmma_f32_16x16x32_bf16(false, a.v, false, bb.v,
                                                      (short)0, acc, false, false);
    }

    const int s = nblk + wn * 16 + l16;      // D: N across lanes -> coalesced stores
    const int ob = m0 + half * 8;            // D: M across VGPRs (+8 for upper half-wave)
    for (int v = 0; v < 8; ++v) {
        int o = ob + v;
        pw[((size_t)b * OC_TOTAL + o) * S_PER_B + s] = (bf16)(acc[v] + bias_all[o]);
    }
}

// ---------------------------------------------------------------------------
// Kernel 3: directional gated tridiagonal scan (4 launches, dir = 0..3).
//   One block per (channel, batch); 64 lanes own the tap dimension, LDS ring
//   buffer carries h across the 64 sequential steps. Tap-normalization and
//   sigmoid fused into the loads; next-step streams prefetched.
// ---------------------------------------------------------------------------
__device__ __forceinline__ int scan_s(int dir, int t, int x) {
    if      (dir == 0) return t * WDIM + x;
    else if (dir == 1) return (HDIM - 1 - t) * WDIM + x;
    else if (dir == 2) return x * WDIM + t;
    else               return x * WDIM + (WDIM - 1 - t);
}

__global__ void k_scan(const bf16* __restrict__ pw, float* __restrict__ y, int dir) {
    __shared__ float hbuf[66];
    const int c = blockIdx.x, b = blockIdx.y, x = threadIdx.x;

    const size_t pb  = (size_t)b * OC_TOTAL * S_PER_B;
    const size_t iu  = pb + (size_t)(OFF_U + c) * S_PER_B;
    const size_t iw0 = pb + (size_t)(OFF_W + (dir * 3 + 0) * DIM + c) * S_PER_B;
    const size_t iw1 = iw0 + (size_t)DIM * S_PER_B;
    const size_t iw2 = iw1 + (size_t)DIM * S_PER_B;
    const size_t il  = pb + (size_t)(OFF_L + dir * DIM + c) * S_PER_B;
    float* yc = y + ((size_t)b * DIM + c) * S_PER_B;

    if (x == 0) { hbuf[0] = 0.0f; hbuf[65] = 0.0f; }
    hbuf[x + 1] = 0.0f;
    __syncthreads();

    for (int t = 0; t < HDIM; ++t) {
        const int s = scan_s(dir, t, x);
        if (t + 1 < HDIM) {                       // prefetch next step's streams
            const int sn = scan_s(dir, t + 1, x);
            __builtin_prefetch(&pw[iu  + sn], 0, 1);
            __builtin_prefetch(&pw[iw0 + sn], 0, 1);
            __builtin_prefetch(&pw[iw1 + sn], 0, 1);
            __builtin_prefetch(&pw[iw2 + sn], 0, 1);
            __builtin_prefetch(&pw[il  + sn], 0, 1);
        }
        float u  = (float)pw[iu  + s];
        float w0 = (float)pw[iw0 + s];
        float w1 = (float)pw[iw1 + s];
        float w2 = (float)pw[iw2 + s];
        float lr = (float)pw[il  + s];
        float inv = 1.0f / (fabsf(w0) + fabsf(w1) + fabsf(w2) + 1e-6f);
        float lam = 1.0f / (1.0f + __expf(-lr));
        float h = (w0 * hbuf[x] + w1 * hbuf[x + 1] + w2 * hbuf[x + 2]) * inv + lam * u;
        __syncthreads();
        hbuf[x + 1] = h;
        __syncthreads();
        yc[s] += h;
    }
}

// ---------------------------------------------------------------------------
// Kernel 4: x1 = x + Wo @ y + bo   (same WMMA tiling; f32 in/out, float4 stage)
// ---------------------------------------------------------------------------
__global__ void __launch_bounds__(256) k_outproj(
        const bf16* __restrict__ wt, const float* __restrict__ y,
        const float* __restrict__ xin, const float* __restrict__ bo,
        float* __restrict__ x1) {
    __shared__ __align__(16) bf16 Bs[32][264];

    const int b = blockIdx.z, mblk = blockIdx.x * 64, nblk = blockIdx.y * 32;
    const int tid = threadIdx.x;

    const float* yb = y + (size_t)b * DIM * S_PER_B;
    for (int idx = tid; idx < DIM * 8; idx += 256) {     // float4 reads along s
        int c = idx >> 3, sq = (idx & 7) * 4;
        float4 f = *(const float4*)(yb + (size_t)c * S_PER_B + nblk + sq);
        Bs[sq + 0][c] = (bf16)f.x;  Bs[sq + 1][c] = (bf16)f.y;
        Bs[sq + 2][c] = (bf16)f.z;  Bs[sq + 3][c] = (bf16)f.w;
    }
    __syncthreads();

    const int wave = tid >> 5, lane = tid & 31;
    const int wm = wave >> 1, wn = wave & 1;
    const int half = lane >> 4, l16 = lane & 15;
    const int m0 = mblk + wm * 16;

    const bf16* Arow = wt + WT_WO + (size_t)(m0 + l16) * DIM;
    const bf16* Brow = &Bs[wn * 16 + l16][0];

    v8f acc = {};
    for (int kb = 0; kb < DIM; kb += 32) {
        V16 a, bb;
        int ka = kb + half * 8;
        a.u[0] = *(const u32x4*)(Arow + ka);
        a.u[1] = *(const u32x4*)(Arow + ka + 16);
        int k2 = kb + half * 8;
        bb.u[0] = *(const u32x4*)(Brow + k2);
        bb.u[1] = *(const u32x4*)(Brow + k2 + 16);
        acc = __builtin_amdgcn_wmma_f32_16x16x32_bf16(false, a.v, false, bb.v,
                                                      (short)0, acc, false, false);
    }

    const int s = nblk + wn * 16 + l16;
    for (int v = 0; v < 8; ++v) {
        int o = m0 + half * 8 + v;
        size_t oi = ((size_t)b * DIM + o) * S_PER_B + s;
        x1[oi] = xin[oi] + bo[o] + acc[v];
    }
}

// ---------------------------------------------------------------------------
// Kernel 5: fused MLP.  out = x1 + gelu(x1 @ W1 + b1) @ W2 + b2
//   Per block: 16-position tile, full channel extent. Phase 1 produces the
//   1024-wide gelu activation directly in LDS in WMMA-B layout (packed 128-bit
//   ds stores); phase 2 consumes it for the 1024->256 down-projection.
// ---------------------------------------------------------------------------
__global__ void __launch_bounds__(256) k_mlp(
        const bf16* __restrict__ wt, const float* __restrict__ x1,
        const float* __restrict__ b1, const float* __restrict__ b2,
        float* __restrict__ out) {
    __shared__ __align__(16) bf16 Bx[16][264];    // x1 tile transposed [s][c]
    __shared__ __align__(16) bf16 Hs[16][1040];   // gelu(h) transposed [s][j]

    const int b = blockIdx.y;
    const int nblk = blockIdx.x * 16;
    const int tid = threadIdx.x;

    const float* xb = x1 + (size_t)b * DIM * S_PER_B;
    for (int idx = tid; idx < DIM * 4; idx += 256) {      // float4 reads along s
        int c = idx >> 2, sq = (idx & 3) * 4;
        float4 f = *(const float4*)(xb + (size_t)c * S_PER_B + nblk + sq);
        Bx[sq + 0][c] = (bf16)f.x;  Bx[sq + 1][c] = (bf16)f.y;
        Bx[sq + 2][c] = (bf16)f.z;  Bx[sq + 3][c] = (bf16)f.w;
    }
    __syncthreads();

    const int wave = tid >> 5, lane = tid & 31;
    const int half = lane >> 4, l16 = lane & 15;

    // ---- phase 1: h[j][s] = gelu(W1t[j,:] . x1[:,s] + b1[j]), 64 M-tiles ----
    for (int mt = wave; mt < 64; mt += 8) {
        const int m0 = mt * 16;
        const bf16* Arow = wt + WT_W1T + (size_t)(m0 + l16) * DIM;
        const bf16* Brow = &Bx[l16][0];
        v8f acc = {};
        for (int kb = 0; kb < DIM; kb += 32) {
            V16 a, bb;
            int ka = kb + half * 8;
            a.u[0] = *(const u32x4*)(Arow + ka);
            a.u[1] = *(const u32x4*)(Arow + ka + 16);
            int k2 = kb + half * 8;
            bb.u[0] = *(const u32x4*)(Brow + k2);
            bb.u[1] = *(const u32x4*)(Brow + k2 + 16);
            acc = __builtin_amdgcn_wmma_f32_16x16x32_bf16(false, a.v, false, bb.v,
                                                          (short)0, acc, false, false);
        }
        const int jb = m0 + half * 8;           // 8 consecutive j per lane -> packed store
        __align__(16) bf16 tmp[8];
        for (int v = 0; v < 8; ++v) {
            float z = acc[v] + b1[jb + v];
            tmp[v] = (bf16)(0.5f * z * (1.0f + erff(z * 0.70710678118f)));  // exact gelu
        }
        *(u32x4*)(&Hs[l16][jb]) = *(const u32x4*)tmp;
    }
    __syncthreads();

    // ---- phase 2: out[o][s] = x1 + W2t[o,:] . h[:,s] + b2[o], 16 M-tiles ----
    for (int mt = wave; mt < 16; mt += 8) {
        const int m0 = mt * 16;
        const bf16* Arow = wt + WT_W2T + (size_t)(m0 + l16) * (4 * DIM);
        const bf16* Brow = &Hs[l16][0];
        v8f acc = {};
        for (int kb = 0; kb < 4 * DIM; kb += 32) {
            V16 a, bb;
            int ka = kb + half * 8;
            a.u[0] = *(const u32x4*)(Arow + ka);
            a.u[1] = *(const u32x4*)(Arow + ka + 16);
            int k2 = kb + half * 8;
            bb.u[0] = *(const u32x4*)(Brow + k2);
            bb.u[1] = *(const u32x4*)(Brow + k2 + 16);
            acc = __builtin_amdgcn_wmma_f32_16x16x32_bf16(false, a.v, false, bb.v,
                                                          (short)0, acc, false, false);
        }
        const int s = nblk + l16;
        for (int v = 0; v < 8; ++v) {
            int o = m0 + half * 8 + v;
            size_t oi = ((size_t)b * DIM + o) * S_PER_B + s;
            out[oi] = x1[oi] + b2[o] + acc[v];
        }
    }
}

// ---------------------------------------------------------------------------
// Launch: prep -> fused pointwise GEMM -> 4 scans -> out-proj -> fused MLP
// ---------------------------------------------------------------------------
extern "C" void kernel_launch(void* const* d_in, const int* in_sizes, int n_in,
                              void* d_out, int out_size, void* d_ws, size_t ws_size,
                              hipStream_t stream) {
    (void)in_sizes; (void)n_in; (void)out_size; (void)ws_size;

    const float* x  = (const float*)d_in[0];
    const float* Wu = (const float*)d_in[1];
    const float* bu = (const float*)d_in[2];
    const float* Ww = (const float*)d_in[3];
    const float* bw = (const float*)d_in[4];
    const float* Wl = (const float*)d_in[5];
    const float* bl = (const float*)d_in[6];
    const float* Wo = (const float*)d_in[7];
    const float* bo = (const float*)d_in[8];
    const float* W1 = (const float*)d_in[9];
    const float* b1 = (const float*)d_in[10];
    const float* W2 = (const float*)d_in[11];
    const float* b2 = (const float*)d_in[12];
    float* out = (float*)d_out;

    // Workspace carve-up (~373 MB total)
    char* ws = (char*)d_ws;
    size_t off = 0;
    bf16*  wt       = (bf16*)(ws + off);  off += (size_t)WT_TOTAL * 2;                   // 3.4 MB
    float* bias_all = (float*)(ws + off); off += (size_t)OC_TOTAL * 4;                   // 17 KB
    bf16*  xbf_t    = (bf16*)(ws + off);  off += (size_t)BATCH * DIM * S_PER_B * 2;      // 16.8 MB
    bf16*  pw       = (bf16*)(ws + off);  off += (size_t)BATCH * OC_TOTAL * S_PER_B * 2; // 285 MB
    float* yb       = (float*)(ws + off); off += (size_t)BATCH * DIM * S_PER_B * 4;      // 33.5 MB
    float* x1       = (float*)(ws + off);                                                // 33.5 MB

    k_prep_weights<<<dim3((12 * DIM * DIM + 255) / 256), 256, 0, stream>>>(
        Wu, Ww, Wl, Wo, W1, W2, bu, bw, bl, wt, bias_all);

    k_prep_x<<<dim3(DIM / 64, S_PER_B / 64, BATCH), 256, 0, stream>>>(x, xbf_t, yb);

    k_pointwise<<<dim3(OC_TOTAL / 64, S_PER_B / 32, BATCH), 256, 0, stream>>>(
        wt, xbf_t, bias_all, pw);

    for (int dir = 0; dir < 4; ++dir)
        k_scan<<<dim3(DIM, BATCH), 64, 0, stream>>>(pw, yb, dir);

    k_outproj<<<dim3(DIM / 64, S_PER_B / 32, BATCH), 256, 0, stream>>>(
        wt, yb, x, bo, x1);

    k_mlp<<<dim3(S_PER_B / 16, BATCH), 256, 0, stream>>>(wt, x1, b1, b2, out);
}
